// LSTM_40776419508630
// MI455X (gfx1250) — matrix operand order repaired and checked by reading, not compiled, not run on hardware
//
#include <hip/hip_runtime.h>

// ---------------------------------------------------------------------------
// Persistent 2-layer LSTM on gfx1250 (MI455X).
//  - 16 workgroups (one WGP each), 256 threads (8 wave32s) per WG.
//  - WG j owns output columns [16j, 16j+16) of every gate, both layers.
//  - Weights: f32 -> bf16, pre-swizzled into the WMMA B-fragment layout,
//    LDS-resident for the whole sequence (128 KB of the 320 KB WGP LDS).
//  - h state lives in global scratch as bf16 *already in WMMA A-fragment
//    layout* (written once by the producer), so WMMA consumes it directly
//    from L2 with coalesced global_load_b128 — no per-step re-swizzle.
//  - Only the x half of layer-0's A operand is staged (float4 -> bf16x4,
//    8 vector iterations/thread) into LDS each step.
//  - One device-scope barrier per timestep (layer0 -> layer1 handoff; the
//    layer1(t) -> layer1(t+1) dependency is covered by step t+1's barrier).
// ---------------------------------------------------------------------------

typedef __bf16  bf16_t;
typedef bf16_t  v16bf __attribute__((ext_vector_type(16)));
typedef float   v8f   __attribute__((ext_vector_type(8)));
typedef float   f4    __attribute__((ext_vector_type(4)));
typedef unsigned short us4 __attribute__((ext_vector_type(4)));

constexpr int kNWG   = 16;
constexpr int kBlock = 256;
constexpr int kB = 32, kT = 2048, kD = 256, kH = 256, kL = 2;
constexpr int kK   = kD + kH;      // 512
constexpr int kNKT = kK / 32;      // 16 K-tiles of 32
constexpr int kNCol = kH / kNWG;   // 16 columns per WG per gate

// output layout: [outputs (T*B,H)][hT (L,B,H)][cT (L,B,H)]
constexpr long long kOutHT = (long long)kT * kB * kH;
constexpr long long kOutCT = kOutHT + (long long)kL * kB * kH;

// ---- dynamic LDS layout (bytes) ----
constexpr int SM_W_CNT  = kL * 4 * kNKT * 32 * 16;   // 65536 ushorts (128 KB)
constexpr int SM_A_CNT  = 2 * 8 * 32 * 16;           // x half only: 8192 ushorts
constexpr int SM_G_CNT  = 4 * kB * 16;               // 2048 floats (8 KB)
constexpr int SM_C_CNT  = kL * kB * 16;              // 1024 floats (4 KB)
constexpr int SM_BIAS_CNT = kL * 4 * 16;             // 128 floats
constexpr int SM_W_OFF  = 0;
constexpr int SM_A_OFF  = SM_W_OFF + SM_W_CNT * 2;
constexpr int SM_G_OFF  = SM_A_OFF + SM_A_CNT * 2;
constexpr int SM_C_OFF  = SM_G_OFF + SM_G_CNT * 4;
constexpr int SM_BIAS_OFF = SM_C_OFF + SM_C_CNT * 4;
constexpr int SM_BYTES  = SM_BIAS_OFF + SM_BIAS_CNT * 4;  // 160256 B

// global scratch: hA = bf16 A-fragments [L][parity][mt(2)][kt(8)][lane(32)][e(16)]
constexpr int kHATileUS = 8 * 32 * 16;               // ushorts per (l,p,mt) block
constexpr int kHAUS     = kL * 2 * 2 * kHATileUS;    // 32768 ushorts = 64 KB

__device__ __forceinline__ unsigned short f2bf(float x) {
  unsigned u = __float_as_uint(x);
  return (unsigned short)((u + 0x7FFFu + ((u >> 16) & 1u)) >> 16);  // RNE
}
__device__ __forceinline__ float sigm(float x) {
  return 1.0f / (1.0f + __expf(-x));
}
__device__ __forceinline__ v8f wmma_bf16(v16bf a, v16bf b, v8f c) {
  return __builtin_amdgcn_wmma_f32_16x16x32_bf16(false, a, false, b,
                                                 (short)0, c, false, false);
}

// Monotonic device-scope barrier across the 16 co-resident WGs.
__device__ __forceinline__ void grid_barrier(unsigned* cnt, unsigned target) {
  __syncthreads();
  if (threadIdx.x == 0) {
    __threadfence();
    __hip_atomic_fetch_add(cnt, 1u, __ATOMIC_ACQ_REL, __HIP_MEMORY_SCOPE_AGENT);
    while (__hip_atomic_load(cnt, __ATOMIC_ACQUIRE, __HIP_MEMORY_SCOPE_AGENT) < target) {
      __builtin_amdgcn_s_sleep(2);
    }
  }
  __syncthreads();
}

// Shared gate -> state epilogue. Writes h as bf16 A-fragments (global) and
// outputs / hT as f32.
__device__ __forceinline__ void gates_to_state(
    int l, int t, int curp, float* __restrict__ out,
    unsigned short* __restrict__ hA,
    const float* smG, float* smC, const float* smBias, int col0)
{
  for (int s = threadIdx.x; s < kB * 16; s += kBlock) {
    int m = s >> 4, n = s & 15;
    float gf = sigm(smG[(0 * kB + m) * 16 + n] + smBias[(l * 4 + 0) * 16 + n]);
    float gi = sigm(smG[(1 * kB + m) * 16 + n] + smBias[(l * 4 + 1) * 16 + n]);
    float gc = tanhf(smG[(2 * kB + m) * 16 + n] + smBias[(l * 4 + 2) * 16 + n]);
    float go = sigm(smG[(3 * kB + m) * 16 + n] + smBias[(l * 4 + 3) * 16 + n]);
    float cn = gf * smC[(l * kB + m) * 16 + n] + gi * gc;
    smC[(l * kB + m) * 16 + n] = cn;
    float hn = go + tanhf(cn);  // faithful to reference: additive

    // scatter into A-fragment slot: K' = col0+n within the 256-wide h block
    int col = col0 + n;
    int ktp = col >> 5, r = col & 31;
    int lh  = (r >> 3) & 1;
    int e   = (r & 7) + ((r >> 4) << 3);
    int lane2 = (m & 15) + (lh << 4);
    hA[((((l * 2 + curp) * 2 + (m >> 4)) * 8 + ktp) * 32 + lane2) * 16 + e] = f2bf(hn);

    if (l == 1) out[((long long)t * kB + m) * kH + col] = hn;
    if (t == kT - 1) out[kOutHT + (l * kB + m) * kH + col] = hn;
  }
}

__global__ __launch_bounds__(kBlock, 1)
void lstm_persistent(const float* __restrict__ x,
                     const float* __restrict__ Wf, const float* __restrict__ bfv,
                     const float* __restrict__ Wi, const float* __restrict__ biv,
                     const float* __restrict__ Wc, const float* __restrict__ bcv,
                     const float* __restrict__ Wo, const float* __restrict__ bov,
                     float* __restrict__ out,
                     unsigned short* __restrict__ hA,   // bf16 h fragments
                     unsigned* __restrict__ barrier_cnt)
{
  extern __shared__ char smem[];
  unsigned short* smW    = (unsigned short*)(smem + SM_W_OFF);
  unsigned short* smA    = (unsigned short*)(smem + SM_A_OFF);
  float*          smG    = (float*)(smem + SM_G_OFF);
  float*          smC    = (float*)(smem + SM_C_OFF);
  float*          smBias = (float*)(smem + SM_BIAS_OFF);

  const int tid  = threadIdx.x;
  const int col0 = blockIdx.x * kNCol;
  const float* Wp[4] = {Wf, Wi, Wc, Wo};
  const float* bp[4] = {bfv, biv, bcv, bov};

  // ---- one-time: weights -> bf16 in WMMA B-fragment layout in LDS ----
  // B-layout (16-bit 32x16): lane<16 -> N=lane, K=e/(e+8); lane>=16: N=lane-16, K base +8.
  for (int s = tid; s < SM_W_CNT; s += kBlock) {
    int e = s & 15, lane = (s >> 4) & 31, kt = (s >> 9) & 15;
    int g = (s >> 13) & 3, l = (s >> 15) & 1;
    int k = kt * 32 + ((lane >> 4) << 3) + (e & 7) + ((e >> 3) << 4);
    int n = col0 + (lane & 15);
    smW[s] = f2bf(Wp[g][((long long)l * kK + k) * kH + n]);
  }
  for (int s = tid; s < SM_BIAS_CNT; s += kBlock) {
    int n = s & 15, g = (s >> 4) & 3, l = (s >> 6) & 1;
    smBias[s] = bp[g][l * kH + col0 + n];
  }
  for (int s = tid; s < SM_C_CNT; s += kBlock) smC[s] = 0.0f;
  __syncthreads();

  const int wave = tid >> 5, lane = tid & 31;
  const int g = wave >> 1, mt = wave & 1;

  unsigned bars = 0;
  for (int t = 0; t < kT; ++t) {
    const int curp = t & 1, prevp = curp ^ 1;

    // ================= layer 0 =================
    // stage x(t) half of A into LDS: float4 -> bf16x4, fragment layout.
    // A-layout (16-bit 16x32): lane<16: M=lane, K=e/(e+8); lane>=16: K base +8.
    __syncthreads();
    for (int sg = tid; sg < 2048; sg += kBlock) {
      int q = sg & 3, ln = (sg >> 2) & 31, kt = (sg >> 7) & 7, m2 = (sg >> 10) & 1;
      int m = m2 * 16 + (ln & 15);
      int Kbase = kt * 32 + ((ln >> 4) << 3);
      int Ks = Kbase + ((q < 2) ? q * 4 : q * 4 + 8);
      f4 v = *(const f4*)(x + ((long long)m * kT + t) * kD + Ks);
      us4 r = {f2bf(v.x), f2bf(v.y), f2bf(v.z), f2bf(v.w)};
      *(us4*)(smA + ((m2 * 8 + kt) * 32 + ln) * 16 + q * 4) = r;
    }
    __syncthreads();

    {
      v8f acc = {0.f, 0.f, 0.f, 0.f, 0.f, 0.f, 0.f, 0.f};
      const unsigned short* Bp = smW + (0 * 4 + g) * kNKT * 512;
      const unsigned short* Ap = smA + mt * 8 * 512;                       // LDS x
      const unsigned short* Hp = hA + ((0 * 2 + prevp) * 2 + mt) * kHATileUS;
#pragma unroll
      for (int kt = 0; kt < 8; ++kt) {
        v16bf a = *(const v16bf*)(Ap + (kt * 32 + lane) * 16);
        v16bf b = *(const v16bf*)(Bp + (kt * 32 + lane) * 16);
        acc = wmma_bf16(a, b, acc);
      }
#pragma unroll
      for (int kt = 0; kt < 8; ++kt) {
        v16bf a = *(const v16bf*)(Hp + (kt * 32 + lane) * 16);             // global
        v16bf b = *(const v16bf*)(Bp + ((8 + kt) * 32 + lane) * 16);
        acc = wmma_bf16(a, b, acc);
      }
#pragma unroll
      for (int r = 0; r < 8; ++r) {
        int row = mt * 16 + ((lane >> 4) << 3) + r;
        smG[(g * kB + row) * 16 + (lane & 15)] = acc[r];
      }
    }
    __syncthreads();
    gates_to_state(0, t, curp, out, hA, smG, smC, smBias, col0);

    ++bars;
    grid_barrier(barrier_cnt, bars * kNWG);

    // prefetch next step's x slab while layer-1 runs
    if (t + 1 < kT) {
      int m2 = tid >> 3, off = (tid & 7) * 32;
      __builtin_prefetch(x + ((long long)m2 * kT + (t + 1)) * kD + off, 0, 1);
    }

    // ================= layer 1 ================= (A entirely from hA/global)
    {
      v8f acc = {0.f, 0.f, 0.f, 0.f, 0.f, 0.f, 0.f, 0.f};
      const unsigned short* Bp = smW + (1 * 4 + g) * kNKT * 512;
      const unsigned short* H0 = hA + ((0 * 2 + curp) * 2 + mt) * kHATileUS;
      const unsigned short* H1 = hA + ((1 * 2 + prevp) * 2 + mt) * kHATileUS;
#pragma unroll
      for (int kt = 0; kt < 8; ++kt) {
        v16bf a = *(const v16bf*)(H0 + (kt * 32 + lane) * 16);
        v16bf b = *(const v16bf*)(Bp + (kt * 32 + lane) * 16);
        acc = wmma_bf16(a, b, acc);
      }
#pragma unroll
      for (int kt = 0; kt < 8; ++kt) {
        v16bf a = *(const v16bf*)(H1 + (kt * 32 + lane) * 16);
        v16bf b = *(const v16bf*)(Bp + ((8 + kt) * 32 + lane) * 16);
        acc = wmma_bf16(a, b, acc);
      }
#pragma unroll
      for (int r = 0; r < 8; ++r) {
        int row = mt * 16 + ((lane >> 4) << 3) + r;
        smG[(g * kB + row) * 16 + (lane & 15)] = acc[r];
      }
    }
    __syncthreads();
    gates_to_state(1, t, curp, out, hA, smG, smC, smBias, col0);
    __syncthreads();
  }

  // ---- epilogue: cT (hT was written during t == T-1) ----
  for (int s = tid; s < kL * kB * 16; s += kBlock) {
    int l = s >> 9, rem = s & 511, m = rem >> 4, n = rem & 15;
    out[kOutCT + (l * kB + m) * kH + col0 + n] = smC[(l * kB + m) * 16 + n];
  }
}

// Zero hA scratch + barrier counter each call (deterministic, capture-safe).
__global__ void init_ws(unsigned* ws, int n) {
  int i = blockIdx.x * blockDim.x + threadIdx.x;
  if (i < n) ws[i] = 0u;
}

extern "C" void kernel_launch(void* const* d_in, const int* in_sizes, int n_in,
                              void* d_out, int out_size, void* d_ws, size_t ws_size,
                              hipStream_t stream) {
  const float* x  = (const float*)d_in[0];
  const float* Wf = (const float*)d_in[1];
  const float* bf = (const float*)d_in[2];
  const float* Wi = (const float*)d_in[3];
  const float* bi = (const float*)d_in[4];
  const float* Wc = (const float*)d_in[5];
  const float* bc = (const float*)d_in[6];
  const float* Wo = (const float*)d_in[7];
  const float* bo = (const float*)d_in[8];
  float* out = (float*)d_out;

  // workspace: hA bf16 fragments (64 KB), then barrier counter
  unsigned short* hA = (unsigned short*)d_ws;
  unsigned* cnt = (unsigned*)((char*)d_ws + (size_t)kHAUS * 2);

  (void)hipFuncSetAttribute(reinterpret_cast<const void*>(lstm_persistent),
                            hipFuncAttributeMaxDynamicSharedMemorySize, SM_BYTES);

  const int ninit = kHAUS / 2 + 4;  // hA (as uints) + counter
  init_ws<<<(ninit + 255) / 256, 256, 0, stream>>>((unsigned*)d_ws, ninit);
  lstm_persistent<<<kNWG, kBlock, SM_BYTES, stream>>>(
      x, Wf, bf, Wi, bi, Wc, bc, Wo, bo, out, hA, cnt);
}
